// TreeCNN_35734127903227
// MI455X (gfx1250) — compile-verified, matrix operand order checked by reading, not run on hardware
//
#include <hip/hip_runtime.h>

typedef __attribute__((ext_vector_type(16))) _Float16 v16h;
typedef __attribute__((ext_vector_type(8)))  _Float16 v8h;
typedef __attribute__((ext_vector_type(4)))  _Float16 h4;
typedef __attribute__((ext_vector_type(8)))  float    v8f;

#define TREES  256
#define N0     524288
#define N1     65536
#define N2     8192
#define DIN    128
#define HIDN   256
#define NOUT   10
#define EPSV   1e-5f

// ---------------- helpers ----------------

__global__ __launch_bounds__(256) void kzero(float* __restrict__ p, int n) {
  int i = blockIdx.x * 256 + threadIdx.x;
  if (i < n) p[i] = 0.f;
}

// contiguous segment_sum over G rows, float4-vectorized; optional f32 / f16 outputs
__global__ __launch_bounds__(256) void kgroupsum(const float* __restrict__ in,
                                                 float* __restrict__ outf,
                                                 _Float16* __restrict__ outh,
                                                 int G, int C, int total4) {
  int i = blockIdx.x * 256 + threadIdx.x;
  if (i >= total4) return;
  int C4 = C >> 2;
  int c  = (i % C4) * 4, p = i / C4;
  const float* src = in + (size_t)p * G * C + c;
  float4 s = {0.f, 0.f, 0.f, 0.f};
  for (int j = 0; j < G; ++j) {
    float4 v = *(const float4*)(src + (size_t)j * C);
    s.x += v.x; s.y += v.y; s.z += v.z; s.w += v.w;
  }
  if (outf) *(float4*)(outf + (size_t)i * 4) = s;
  if (outh) {
    h4 o; o[0] = (_Float16)s.x; o[1] = (_Float16)s.y;
          o[2] = (_Float16)s.z; o[3] = (_Float16)s.w;
    *(h4*)(outh + (size_t)i * 4) = o;
  }
}

// per-column sum/sumsq over M rows of Z[M][256]; float4 rows, atomics into stats[2*256]
__global__ __launch_bounds__(256) void kcolstats(const float* __restrict__ Z,
                                                 float* __restrict__ stats, int M) {
  int c    = (threadIdx.x & 63) * 4;
  int rsub = threadIdx.x >> 6;                    // 0..3
  int r0   = blockIdx.x * 512 + rsub;
  int r1   = blockIdx.x * 512 + 512; if (r1 > M) r1 = M;
  float4 s = {0.f,0.f,0.f,0.f}, q = {0.f,0.f,0.f,0.f};
  for (int r = r0; r < r1; r += 4) {
    float4 v = *(const float4*)(Z + (size_t)r * HIDN + c);
    s.x += v.x; s.y += v.y; s.z += v.z; s.w += v.w;
    q.x += v.x*v.x; q.y += v.y*v.y; q.z += v.z*v.z; q.w += v.w*v.w;
  }
  atomicAdd(&stats[c+0], s.x); atomicAdd(&stats[c+1], s.y);
  atomicAdd(&stats[c+2], s.z); atomicAdd(&stats[c+3], s.w);
  atomicAdd(&stats[HIDN+c+0], q.x); atomicAdd(&stats[HIDN+c+1], q.y);
  atomicAdd(&stats[HIDN+c+2], q.z); atomicAdd(&stats[HIDN+c+3], q.w);
}

// fused BN(batch stats) + ReLU + G-row pooling; float4; optional f32 / f16 outputs
__global__ __launch_bounds__(256) void kbnpool(const float* __restrict__ Z,
                                               const float* __restrict__ stats,
                                               const float* __restrict__ gam,
                                               const float* __restrict__ bet,
                                               float* __restrict__ outf,
                                               _Float16* __restrict__ outh,
                                               int M, int G, int total4) {
  int i = blockIdx.x * 256 + threadIdx.x;
  if (i >= total4) return;
  int c = (i & 63) * 4, p = i >> 6;
  float invM = 1.f / (float)M;
  float mn[4], sc[4], bb[4];
  #pragma unroll
  for (int e = 0; e < 4; ++e) {
    mn[e] = stats[c+e] * invM;
    float var = stats[HIDN+c+e] * invM - mn[e]*mn[e];
    sc[e] = rsqrtf(var + EPSV) * gam[c+e];
    bb[e] = bet[c+e];
  }
  const float* src = Z + (size_t)p * G * HIDN + c;
  float a[4] = {0.f,0.f,0.f,0.f};
  for (int j = 0; j < G; ++j) {
    float4 v = *(const float4*)(src + (size_t)j * HIDN);
    a[0] += fmaxf((v.x - mn[0]) * sc[0] + bb[0], 0.f);
    a[1] += fmaxf((v.y - mn[1]) * sc[1] + bb[1], 0.f);
    a[2] += fmaxf((v.z - mn[2]) * sc[2] + bb[2], 0.f);
    a[3] += fmaxf((v.w - mn[3]) * sc[3] + bb[3], 0.f);
  }
  if (outf) { float4 o = {a[0],a[1],a[2],a[3]}; *(float4*)(outf + (size_t)i*4) = o; }
  if (outh) {
    h4 o; o[0]=(_Float16)a[0]; o[1]=(_Float16)a[1]; o[2]=(_Float16)a[2]; o[3]=(_Float16)a[3];
    *(h4*)(outh + (size_t)i * 4) = o;
  }
}

// swizzle f32 W[K x 256] -> f16 fragment-ordered buffer.
// element j = ((step*16 + tile)*32 + lane)*16 + slot ; k = step*32 + (lane>>4)*8 + (slot&7) + 16*(slot>>3)
// n = tile*16 + (lane&15)
__global__ __launch_bounds__(256) void kswizw(const float* __restrict__ W,
                                              _Float16* __restrict__ out, int K) {
  int j = blockIdx.x * 256 + threadIdx.x;
  if (j >= K * HIDN) return;
  int i  = j & 15;
  int l  = (j >> 4) & 31;
  int tl = (j >> 9) & 15;
  int s  = j >> 13;
  int k  = s * 32 + ((l >> 4) * 8) + (i & 7) + ((i >> 3) << 4);
  int n  = tl * 16 + (l & 15);
  out[j] = (_Float16)W[(size_t)k * HIDN + n];
}

// ---------------- WMMA GEMM: C[M x 256] = A[M x K](f16,row-major) @ Wf(fragment-order) + bias
// block = 256 threads (8 waves), tile 64 x 256; wave -> (32-row half, 64-col group)
// A fragments: two aligned 16B loads per 16-row tile; B fragments: one aligned 32B load.
__global__ __launch_bounds__(256) void kgemm(const _Float16* __restrict__ A,
                                             const _Float16* __restrict__ Wf,
                                             const float* __restrict__ bias,
                                             float* __restrict__ C,
                                             int M, int K) {
  const int tid   = threadIdx.x;
  const int lane  = tid & 31;
  const int wave  = tid >> 5;
  const int mbase = blockIdx.x * 64 + (wave & 1) * 32;
  const int nb    = (wave >> 1) * 64;
  const int koff  = (lane >> 4) * 8;
  const int lrow  = lane & 15;
  const int r0    = mbase + lrow;
  const v16h* Bv  = (const v16h*)Wf;           // element index = T*32 + lane

  v8f acc[2][4] = {};

  for (int k0 = 0; k0 < K; k0 += 32) {
    v16h a[2];
    #pragma unroll
    for (int u = 0; u < 2; ++u) {
      const _Float16* ap = A + (size_t)(r0 + u * 16) * K + k0 + koff;
      v8h lo = *(const v8h*)ap;
      v8h hi = *(const v8h*)(ap + 16);
      a[u] = __builtin_shufflevector(lo, hi, 0,1,2,3,4,5,6,7,8,9,10,11,12,13,14,15);
    }
    if (k0 + 32 < K)
      __builtin_prefetch(A + (size_t)r0 * K + k0 + 32 + koff, 0, 1);

    const int Tb = (k0 >> 5) * 16 + (nb >> 4);
    #pragma unroll
    for (int t = 0; t < 4; ++t) {
      v16h b = Bv[(size_t)(Tb + t) * 32 + lane];
      #pragma unroll
      for (int u = 0; u < 2; ++u)
        acc[u][t] = __builtin_amdgcn_wmma_f32_16x16x32_f16(
            false, a[u], false, b, (short)0, acc[u][t], false, false);
    }
  }

  // C/D layout: VGPR r -> row +r (lanes 0-15) / +r+8 (lanes 16-31), col = lane&15
  #pragma unroll
  for (int u = 0; u < 2; ++u) {
    const int rb = mbase + u * 16 + ((lane >> 4) << 3);
    #pragma unroll
    for (int t = 0; t < 4; ++t) {
      int col  = nb + t * 16 + lrow;
      float bv = bias[col];
      #pragma unroll
      for (int r = 0; r < 8; ++r)
        C[(size_t)(rb + r) * HIDN + col] = acc[u][t][r] + bv;
    }
  }
}

// ---------------- readout: score[256 x 10] ----------------
__global__ __launch_bounds__(256) void kscore(const float* __restrict__ s0,
                                              const float* __restrict__ s1,
                                              const float* __restrict__ s2,
                                              const float* __restrict__ h3,
                                              const float* __restrict__ p0w,
                                              const float* __restrict__ p0b,
                                              const float* __restrict__ phw,
                                              const float* __restrict__ phb,
                                              float* __restrict__ out) {
  int t = threadIdx.x;   // tree
  int o = blockIdx.x;    // output channel
  float acc = p0b[o] + phb[o] + phb[10 + o] + phb[20 + o];
  for (int k = 0; k < DIN;  ++k) acc += s0[t * DIN  + k] * p0w[k * NOUT + o];
  for (int k = 0; k < HIDN; ++k) acc += s1[t * HIDN + k] * phw[k * NOUT + o];
  for (int k = 0; k < HIDN; ++k) acc += s2[t * HIDN + k] * phw[2560 + k * NOUT + o];
  for (int k = 0; k < HIDN; ++k) acc += h3[t * HIDN + k] * phw[5120 + k * NOUT + o];
  out[t * NOUT + o] = acc;
}

// ---------------- host launcher ----------------
extern "C" void kernel_launch(void* const* d_in, const int* in_sizes, int n_in,
                              void* d_out, int out_size, void* d_ws, size_t ws_size,
                              hipStream_t stream) {
  (void)in_sizes; (void)n_in; (void)out_size; (void)ws_size;
  const float* x       = (const float*)d_in[0];
  const float* l1_w1   = (const float*)d_in[7];
  const float* l1_b1   = (const float*)d_in[8];
  const float* l1_bn1g = (const float*)d_in[9];
  const float* l1_bn1b = (const float*)d_in[10];
  const float* l1_w2   = (const float*)d_in[11];
  const float* l1_b2   = (const float*)d_in[12];
  const float* l1_bng  = (const float*)d_in[13];
  const float* l1_bnb  = (const float*)d_in[14];
  const float* lw1     = (const float*)d_in[15];
  const float* lb1     = (const float*)d_in[16];
  const float* lbn1g   = (const float*)d_in[17];
  const float* lbn1b   = (const float*)d_in[18];
  const float* lw2     = (const float*)d_in[19];
  const float* lb2     = (const float*)d_in[20];
  const float* lbng    = (const float*)d_in[21];
  const float* lbnb    = (const float*)d_in[22];
  const float* p0w     = (const float*)d_in[23];
  const float* p0b     = (const float*)d_in[24];
  const float* phw     = (const float*)d_in[25];
  const float* phb     = (const float*)d_in[26];
  float* out = (float*)d_out;

  char* ws = (char*)d_ws;
  size_t off = 0;
  auto carve = [&](size_t bytes) -> void* {
    void* p = ws + off; off += (bytes + 255) & ~(size_t)255; return p;
  };
  float*    P1f  = (float*)   carve((size_t)N1 * DIN  * 4);
  _Float16* P1h  = (_Float16*)carve((size_t)N1 * DIN  * 2);
  float*    Z    = (float*)   carve((size_t)N1 * HIDN * 4);
  _Float16* T16  = (_Float16*)carve((size_t)N1 * HIDN * 2);
  float*    P2f  = (float*)   carve((size_t)N2 * HIDN * 4);
  _Float16* P2h  = (_Float16*)carve((size_t)N2 * HIDN * 2);
  float*    P3f  = (float*)   carve((size_t)TREES * HIDN * 4);   // == s2
  _Float16* P3h  = (_Float16*)carve((size_t)TREES * HIDN * 2);
  float*    H3   = (float*)   carve((size_t)TREES * HIDN * 4);
  float*    S0   = (float*)   carve((size_t)TREES * DIN  * 4);
  float*    S1   = (float*)   carve((size_t)TREES * HIDN * 4);
  float*    ST   = (float*)   carve(2 * HIDN * 4);
  _Float16* WA   = (_Float16*)carve((size_t)DIN  * HIDN * 2);    // l1_w1 (swizzled)
  _Float16* WB   = (_Float16*)carve((size_t)HIDN * HIDN * 2);    // l1_w2
  _Float16* WC   = (_Float16*)carve((size_t)2 * HIDN * HIDN * 2);// lw1[0..1]
  _Float16* WD   = (_Float16*)carve((size_t)2 * HIDN * HIDN * 2);// lw2[0..1]

  dim3 blk(256);
  auto g = [](int total) { return dim3((unsigned)((total + 255) / 256)); };
  const int WSL = HIDN * HIDN;   // elements per stacked weight slice

  // weight swizzle+convert (fragment order)
  kswizw<<<g(DIN * HIDN),  blk, 0, stream>>>(l1_w1,       WA,        DIN);
  kswizw<<<g(WSL),         blk, 0, stream>>>(l1_w2,       WB,        HIDN);
  kswizw<<<g(WSL),         blk, 0, stream>>>(lw1,         WC,        HIDN);
  kswizw<<<g(WSL),         blk, 0, stream>>>(lw1 + WSL,   WC + WSL,  HIDN);
  kswizw<<<g(WSL),         blk, 0, stream>>>(lw2,         WD,        HIDN);
  kswizw<<<g(WSL),         blk, 0, stream>>>(lw2 + WSL,   WD + WSL,  HIDN);

  // ---- layer 1 ----
  kgroupsum<<<g(N1*DIN/4), blk, 0, stream>>>(x, P1f, P1h, 8, DIN, N1*DIN/4);        // pooled1 (+f16)
  kgroupsum<<<g(TREES*DIN/4), blk, 0, stream>>>(P1f, S0, (_Float16*)nullptr,
                                                256, DIN, TREES*DIN/4);              // s0
  kgemm<<<dim3(N1/64), blk, 0, stream>>>(P1h, WA, l1_b1, Z, N1, DIN);
  kzero<<<g(2*HIDN), blk, 0, stream>>>(ST, 2*HIDN);
  kcolstats<<<dim3((N1+511)/512), blk, 0, stream>>>(Z, ST, N1);
  kbnpool<<<g(N1*64), blk, 0, stream>>>(Z, ST, l1_bn1g, l1_bn1b,
                                        (float*)nullptr, T16, N1, 1, N1*64);
  kgemm<<<dim3(N1/64), blk, 0, stream>>>(T16, WB, l1_b2, Z, N1, HIDN);
  kzero<<<g(2*HIDN), blk, 0, stream>>>(ST, 2*HIDN);
  kcolstats<<<dim3((N1+511)/512), blk, 0, stream>>>(Z, ST, N1);
  kbnpool<<<g(N2*64), blk, 0, stream>>>(Z, ST, l1_bng, l1_bnb,
                                        P2f, P2h, N1, 8, N2*64);                     // pooled2
  kgroupsum<<<g(TREES*HIDN/4), blk, 0, stream>>>(P2f, S1, (_Float16*)nullptr,
                                                 32, HIDN, TREES*HIDN/4);            // s1

  // ---- layer 2 ----
  kgemm<<<dim3(N2/64), blk, 0, stream>>>(P2h, WC, lb1, Z, N2, HIDN);
  kzero<<<g(2*HIDN), blk, 0, stream>>>(ST, 2*HIDN);
  kcolstats<<<dim3((N2+511)/512), blk, 0, stream>>>(Z, ST, N2);
  kbnpool<<<g(N2*64), blk, 0, stream>>>(Z, ST, lbn1g, lbn1b,
                                        (float*)nullptr, T16, N2, 1, N2*64);
  kgemm<<<dim3(N2/64), blk, 0, stream>>>(T16, WD, lb2, Z, N2, HIDN);
  kzero<<<g(2*HIDN), blk, 0, stream>>>(ST, 2*HIDN);
  kcolstats<<<dim3((N2+511)/512), blk, 0, stream>>>(Z, ST, N2);
  kbnpool<<<g(TREES*64), blk, 0, stream>>>(Z, ST, lbng, lbnb,
                                           P3f, P3h, N2, 32, TREES*64);              // pooled3 = s2

  // ---- layer 3 ----
  kgemm<<<dim3(TREES/64), blk, 0, stream>>>(P3h, WC + WSL, lb1 + HIDN, Z, TREES, HIDN);
  kzero<<<g(2*HIDN), blk, 0, stream>>>(ST, 2*HIDN);
  kcolstats<<<dim3(1), blk, 0, stream>>>(Z, ST, TREES);
  kbnpool<<<g(TREES*64), blk, 0, stream>>>(Z, ST, lbn1g + HIDN, lbn1b + HIDN,
                                           (float*)nullptr, T16, TREES, 1, TREES*64);
  kgemm<<<dim3(TREES/64), blk, 0, stream>>>(T16, WD + WSL, lb2 + HIDN, Z, TREES, HIDN);
  kzero<<<g(2*HIDN), blk, 0, stream>>>(ST, 2*HIDN);
  kcolstats<<<dim3(1), blk, 0, stream>>>(Z, ST, TREES);
  kbnpool<<<g(TREES*64), blk, 0, stream>>>(Z, ST, lbng + HIDN, lbnb + HIDN,
                                           H3, (_Float16*)nullptr, TREES, 1, TREES*64);

  // ---- readout ----
  kscore<<<dim3(NOUT), blk, 0, stream>>>(S0, S1, P3f, H3, p0w, p0b, phw, phb, out);
}
